// GaussianKernelLayer_67224828117757
// MI455X (gfx1250) — compile-verified, matrix operand order checked by reading, not run on hardware
//
#include <hip/hip_runtime.h>

typedef __attribute__((ext_vector_type(4)))  unsigned int u32x4;
typedef __attribute__((ext_vector_type(8)))  float        f32x8;
typedef __attribute__((ext_vector_type(16))) __bf16       bf16x16;

#define C_DIM 128
#define T_DIM 512
#define F_DIM 32
#define N_DIM 32
#define LDS_STRIDE 520                       // halves per row; 1040B = 16B aligned, pad breaks bank conflicts
#define LDS_BYTES (C_DIM * LDS_STRIDE * 2 + C_DIM * 4)

// ---------- round-to-nearest-even f32 -> bf16, packed pair ----------
__device__ inline unsigned bf16pack(float a, float b) {
    unsigned ua = __float_as_uint(a);
    unsigned ub = __float_as_uint(b);
    ua = (ua + 0x7FFFu + ((ua >> 16) & 1u)) >> 16;
    ub = (ub + 0x7FFFu + ((ub >> 16) & 1u)) >> 16;
    return (ub << 16) | (ua & 0xFFFFu);
}

// ---------------------------------------------------------------------------
// Kernel 1: (N,C,T,F) f32  ->  xb[(n*F+f)][c][t] bf16 (t contiguous)
// 32(t) x 32(f) tile transpose per (n, c, t-block); coalesced both ways.
// ---------------------------------------------------------------------------
__global__ void __launch_bounds__(256)
transpose_pack_kernel(const float* __restrict__ x, unsigned short* __restrict__ xb) {
    __shared__ float tile[32][33];
    const int blk = blockIdx.x;            // ((n*C + c)*16 + tb)
    const int tb = blk & 15;
    const int c  = (blk >> 4) & (C_DIM - 1);
    const int n  = blk >> 11;
    const int t0 = tb * 32;

    const int f  = threadIdx.x & 31;
    const int i0 = threadIdx.x >> 5;       // 0..7
    const size_t gbase = ((size_t)(n * C_DIM + c) * T_DIM) * F_DIM;
#pragma unroll
    for (int p = 0; p < 4; ++p) {
        const int i = i0 + p * 8;
        tile[i][f] = x[gbase + (size_t)(t0 + i) * F_DIM + f];   // contiguous in f
    }
    __syncthreads();

    const int j2 = (threadIdx.x & 15) * 2; // t pair within tile
    const int fq = threadIdx.x >> 4;       // 0..15
    unsigned* __restrict__ xb32 = (unsigned*)xb;
#pragma unroll
    for (int p = 0; p < 2; ++p) {
        const int fi = fq + p * 16;
        const unsigned v = bf16pack(tile[j2][fi], tile[j2 + 1][fi]);
        const unsigned idx = (((unsigned)(n * F_DIM + fi) * C_DIM + c) * T_DIM + t0 + j2) >> 1;
        xb32[idx] = v;                     // 4B dwords, contiguous in t
    }
}

// ---------------------------------------------------------------------------
// Fragment load from LDS, row-major [C][LDS_STRIDE] bf16.
// A(16x32) and B(32x16) bf16 WMMA layouts are the same per-lane pattern:
//   lane<16 : row = rb*16+lane,    K = k0+{0..7} (v0..3), k0+{16..23} (v4..7)
//   lane>=16: row = rb*16+lane-16, K = k0+{8..15},        k0+{24..31}
// ---------------------------------------------------------------------------
__device__ inline bf16x16 load_frag(const unsigned short* Xs, int rb, int lrow, int hi, int k0) {
    const unsigned short* p = Xs + (rb * 16 + lrow) * LDS_STRIDE + k0 + hi * 8;
    struct P { u32x4 a, b; } pp;
    pp.a = *(const u32x4*)(p);             // 16B: 8 halves
    pp.b = *(const u32x4*)(p + 16);        // 16B: 8 halves, +32B
    return __builtin_bit_cast(bf16x16, pp);
}

// ---------------------------------------------------------------------------
// Kernel 2: one workgroup per batch b = n*F + f.  X(128x512 bf16) in LDS,
// full 8x8 grid of 16x16 WMMA tiles, diag from Gram diagonal, exp epilogue.
// Wave w owns row-block w. Per K-step: 1 A frag + ALL 8 B frags into distinct
// registers (sched_barrier keeps the loads grouped ahead of the WMMAs so the
// dscnt waits are staged instead of 8x full-latency stalls).
// ---------------------------------------------------------------------------
__global__ void __launch_bounds__(256)
gram_wmma_kernel(const unsigned short* __restrict__ xb,
                 const float* __restrict__ sigma,
                 float* __restrict__ out) {
    extern __shared__ char smem[];
    unsigned short* Xs = (unsigned short*)smem;
    float* diag = (float*)(smem + C_DIM * LDS_STRIDE * 2);

    const int b    = blockIdx.x;
    const int tid  = threadIdx.x;
    // wave id is uniform across the wave: pin it in an SGPR so guards below
    // compile to scalar branches (no exec-mask tree, no duplicated blocks).
    const int w    = __builtin_amdgcn_readfirstlane(tid >> 5);   // wave 0..7
    const int lane = tid & 31;
    const int hi   = lane >> 4;
    const int lrow = lane & 15;

    // ---- Phase 0: stage X[b] (128KB bf16) into LDS, b128 coalesced ----
    {
        const u32x4* __restrict__ src = (const u32x4*)(xb + (size_t)b * (C_DIM * T_DIM));
#pragma unroll
        for (int it = 0; it < 32; ++it) {
            const int chunk = tid + it * 256;          // 0..8191 (16B chunks)
            const int c = chunk >> 6;                  // 64 chunks per row
            const int j = chunk & 63;
            *(u32x4*)(Xs + c * LDS_STRIDE + j * 8) = src[chunk];
        }
    }
    const float sg = sigma[0];
    const float sc = -0.5f / (sg * sg);
    __syncthreads();

    // ---- Phase 1: wave w computes tiles (w, 0..7); K = 512 in steps of 32 ----
    f32x8 acc[8];
#pragma unroll
    for (int t = 0; t < 8; ++t) acc[t] = f32x8{0, 0, 0, 0, 0, 0, 0, 0};

#pragma unroll 2
    for (int k0 = 0; k0 < T_DIM; k0 += 32) {
        const bf16x16 fa = load_frag(Xs, w, lrow, hi, k0);
        bf16x16 fb[8];
#pragma unroll
        for (int tj = 0; tj < 8; ++tj)
            fb[tj] = load_frag(Xs, tj, lrow, hi, k0);   // 16 ds_load_b128, distinct regs
        __builtin_amdgcn_sched_barrier(0);              // keep all loads ahead of WMMAs
#pragma unroll
        for (int tj = 0; tj < 8; ++tj)
            acc[tj] = __builtin_amdgcn_wmma_f32_16x16x32_bf16(
                false, fa, false, fb[tj], (short)0, acc[tj], false, false);
    }

    // ---- Phase 2: publish Gram diagonal (tile (w,w) lives in acc[w]) ----
    // Compile-time t + wave-uniform scalar guard (t == w): acc[] only ever
    // indexed by constants -> stays in registers, and the guard is an s_cbranch.
#pragma unroll
    for (int t = 0; t < 8; ++t) {
        if (t == w) {
#pragma unroll
            for (int r = 0; r < 8; ++r) {
                const int row_loc = r + hi * 8;
                if (lrow == row_loc) diag[t * 16 + row_loc] = acc[t][r];
            }
        }
    }
    __syncthreads();

    // ---- Phase 3: d2 = Gii + Gjj - 2 Gij ; kern = exp(d2 * sc) ----
    float di[8];
#pragma unroll
    for (int r = 0; r < 8; ++r) di[r] = diag[w * 16 + r + hi * 8];

    const int n = b >> 5;                  // b = n*F + f
    const int f = b & 31;
#pragma unroll
    for (int tj = 0; tj < 8; ++tj) {
        const float dj = diag[tj * 16 + lrow];
        const int cj = tj * 16 + lrow;
#pragma unroll
        for (int r = 0; r < 8; ++r) {
            const int ci = w * 16 + r + hi * 8;
            float d2 = di[r] + dj - 2.0f * acc[tj][r];
            d2 = fmaxf(d2, 0.0f);
            out[((unsigned)(n * C_DIM + ci) * C_DIM + cj) * F_DIM + f] = __expf(d2 * sc);
        }
    }
}

extern "C" void kernel_launch(void* const* d_in, const int* in_sizes, int n_in,
                              void* d_out, int out_size, void* d_ws, size_t ws_size,
                              hipStream_t stream) {
    const float* x     = (const float*)d_in[0];
    const float* sigma = (const float*)d_in[1];
    float* out = (float*)d_out;
    unsigned short* xb = (unsigned short*)d_ws;   // 1024*128*512*2 = 128 MiB scratch

    (void)in_sizes; (void)n_in; (void)out_size; (void)ws_size;
    static_cast<void>(hipFuncSetAttribute((const void*)gram_wmma_kernel,
                                          hipFuncAttributeMaxDynamicSharedMemorySize,
                                          (int)LDS_BYTES));

    transpose_pack_kernel<<<N_DIM * C_DIM * (T_DIM / 32), 256, 0, stream>>>(x, xb);
    gram_wmma_kernel<<<N_DIM * F_DIM, 256, LDS_BYTES, stream>>>(xb, sigma, out);
}
